// MHCLayer_33148557590893
// MI455X (gfx1250) — matrix operand order repaired and checked by reading, not compiled, not run on hardware
//
#include <hip/hip_runtime.h>
#include <hip/hip_bf16.h>
#include <math.h>

typedef __attribute__((ext_vector_type(16))) _Float16 v16h;
typedef __attribute__((ext_vector_type(8)))  _Float16 v8h;
typedef __attribute__((ext_vector_type(8)))  float    v8f;

#define DIMC   1024
#define HIDN   4096
#define NTOK   32                  // tokens per block (2 WMMA M-tiles)
#define MT     2                   // M-tiles per block
#define KCH    512                 // GEMM1 output-column chunk == GEMM2 K chunk
#define NCHUNK (HIDN / KCH)        // 8
#define NGATE  4

// ---------------- A operand from LDS (16-bit A, 16x32, wave32 layout) ----------------
// lane<16 : row M=lane,    halves 0..7 -> K=kb+0..7,  halves 8..15 -> K=kb+16..23
// lane>=16: row M=lane-16, halves 0..7 -> K=kb+8..15, halves 8..15 -> K=kb+24..31
__device__ __forceinline__ v16h load_a_lds(const _Float16* base, int ld, int lane, int kb) {
  const int row = lane & 15;
  const int off = (lane & 16) >> 1;                  // 0 or 8
  const _Float16* p = base + row * ld + kb + off;
  v8h g0 = *(const v8h*)(p);
  v8h g1 = *(const v8h*)(p + 16);
  return __builtin_shufflevector(g0, g1, 0,1,2,3,4,5,6,7,8,9,10,11,12,13,14,15);
}

// ---------------- B operand from global (16-bit B, 32x16, wave32 layout) -------------
// lane holds column N=lane%16; halves 0..7 -> K = kb + (lane>=16 ? 8 : 0) + 0..7, halves 8..15 -> K = kb + 16 + (lane>=16 ? 8 : 0) + 0..7
// B[k][n] = W[row n][k] (row-major weight, reduction along the row) -> two 16B loads.
__device__ __forceinline__ v16h load_b_glb(const _Float16* base, size_t ld, int nb, int kb, int lane) {
  const int col  = lane & 15;
  const int koff = (lane & 16) >> 1;                 // 0 or 8
  const _Float16* p = base + (size_t)(nb + col) * ld + (size_t)(kb + koff);
  v8h g0 = *(const v8h*)(p);
  v8h g1 = *(const v8h*)(p + 16);
  return __builtin_shufflevector(g0, g1, 0,1,2,3,4,5,6,7,8,9,10,11,12,13,14,15);
}

__device__ __forceinline__ float sigmoidf_(float z) { return 1.f / (1.f + expf(-z)); }

// ================= prep: fold rms_w into W_proj, convert W1/W2 to f16 =================
__global__ __launch_bounds__(256) void mhc_prep(
    const float* __restrict__ W_proj, const float* __restrict__ rms_w,
    const float* __restrict__ W1, const float* __restrict__ W2,
    float* __restrict__ W_eff, _Float16* __restrict__ W1h, _Float16* __restrict__ W2h)
{
  const int tid = blockIdx.x * blockDim.x + threadIdx.x;
  if (tid < 24 * DIMC) {
    const int k = tid / DIMC, c = tid % DIMC;
    float s = 0.f;
    #pragma unroll
    for (int i = 0; i < NGATE; ++i)
      s = fmaf(W_proj[(size_t)k * (NGATE * DIMC) + i * DIMC + c], rms_w[i * DIMC + c], s);
    W_eff[tid] = s;
  }
  const int total = HIDN * DIMC;
  const int stride = gridDim.x * blockDim.x;
  for (int i = tid; i < total; i += stride) {
    W1h[i] = (_Float16)W1[i];
    W2h[i] = (_Float16)W2[i];
  }
}

// ================================== main fused kernel =================================
__global__ __launch_bounds__(256, 1) void mhc_main(
    const float* __restrict__ x,
    const float* __restrict__ W_eff,
    const float* __restrict__ bias_pre, const float* __restrict__ bias_post,
    const float* __restrict__ bias_res,
    const float* __restrict__ alpha_pre, const float* __restrict__ alpha_post,
    const float* __restrict__ alpha_res,
    const _Float16* __restrict__ W1h, const float* __restrict__ b1,
    const _Float16* __restrict__ W2h, const float* __restrict__ b2,
    float* __restrict__ out)
{
  __shared__ _Float16 xh[NTOK * DIMC];      // 64KB: f16 raw x tile (GEMM1 A operand)
  __shared__ _Float16 hch[NTOK * KCH];      // 32KB: f16 GELU chunk (GEMM2 A operand)
  __shared__ float s_rms[NTOK];
  __shared__ float s_inv[NTOK];
  __shared__ float s_dyn[NTOK * 24];
  __shared__ float s_pre[NTOK];             // sum(H_pre) per token
  __shared__ float s_post[NTOK * NGATE];    // H_post
  __shared__ float s_res[NTOK * NGATE];     // row sums of sinkhorn matrix

  const int tid  = threadIdx.x;
  const int lane = tid & 31;
  const int wv   = tid >> 5;                // wave index 0..7
  const size_t tok0 = (size_t)blockIdx.x * NTOK;

  if (tid < NTOK) s_rms[tid] = 0.f;
  __syncthreads();

  // ---- load x tile -> f16 LDS, accumulate sum of squares per token ----
  {
    const int row = tid >> 3;               // 8 threads per token row (32 rows)
    const int sub = tid & 7;
    const float4* xr = (const float4*)(x + (tok0 + row) * DIMC);
    float ss = 0.f;
    #pragma unroll 4
    for (int v = sub; v < DIMC / 4; v += 8) {
      float4 f = xr[v];
      ss = fmaf(f.x, f.x, fmaf(f.y, f.y, fmaf(f.z, f.z, fmaf(f.w, f.w, ss))));
      _Float16* d = xh + row * DIMC + v * 4;
      d[0] = (_Float16)f.x; d[1] = (_Float16)f.y;
      d[2] = (_Float16)f.z; d[3] = (_Float16)f.w;
    }
    atomicAdd(&s_rms[row], ss);
  }
  __syncthreads();
  // mean(x_flat^2) over 4*DIMC of 4 tiled copies == sum(x^2)/DIMC
  if (tid < NTOK) s_inv[tid] = 1.f / sqrtf(s_rms[tid] * (1.f / (float)DIMC) + 1e-8f);
  __syncthreads();

  // ---- dyn = (x/rms) . W_eff  (24 gates per token; folded rms_w) ----
  for (int task = tid; task < NTOK * 24; task += 256) {   // 768 tasks, 3 per thread
    const int t = task / 24, k = task % 24;
    const _Float16* xr = xh + t * DIMC;
    const float* w = W_eff + k * DIMC;
    float s = 0.f;
    #pragma unroll 2
    for (int c = 0; c < DIMC; c += 8) {
      v8h xv = *(const v8h*)(xr + c);
      const float4 w0 = *(const float4*)(w + c);
      const float4 w1 = *(const float4*)(w + c + 4);
      s = fmaf((float)xv[0], w0.x, s); s = fmaf((float)xv[1], w0.y, s);
      s = fmaf((float)xv[2], w0.z, s); s = fmaf((float)xv[3], w0.w, s);
      s = fmaf((float)xv[4], w1.x, s); s = fmaf((float)xv[5], w1.y, s);
      s = fmaf((float)xv[6], w1.z, s); s = fmaf((float)xv[7], w1.w, s);
    }
    s_dyn[t * 24 + k] = s * s_inv[t];
  }
  __syncthreads();

  // ---- gates: sigmoid / 2*sigmoid / sinkhorn (one thread per token) ----
  if (tid < NTOK) {
    const int t = tid;
    const float ap = alpha_pre[0], apo = alpha_post[0], ar = alpha_res[0];
    const float* d = s_dyn + t * 24;
    float sp = 0.f;
    #pragma unroll
    for (int n = 0; n < NGATE; ++n) sp += sigmoidf_(fmaf(ap, d[n], bias_pre[n]));
    s_pre[t] = sp;
    #pragma unroll
    for (int n = 0; n < NGATE; ++n)
      s_post[t * NGATE + n] = 2.f * sigmoidf_(fmaf(apo, d[NGATE + n], bias_post[n]));
    float M[NGATE][NGATE];
    #pragma unroll
    for (int i = 0; i < NGATE; ++i)
      #pragma unroll
      for (int j = 0; j < NGATE; ++j)
        M[i][j] = expf(fmaf(ar, d[2 * NGATE + i * NGATE + j], bias_res[i * NGATE + j]));
    #pragma unroll
    for (int it = 0; it < 3; ++it) {
      #pragma unroll
      for (int j = 0; j < NGATE; ++j) {
        float cs = M[0][j] + M[1][j] + M[2][j] + M[3][j] + 1e-8f;
        #pragma unroll
        for (int i = 0; i < NGATE; ++i) M[i][j] /= cs;
      }
      #pragma unroll
      for (int i = 0; i < NGATE; ++i) {
        float rs = M[i][0] + M[i][1] + M[i][2] + M[i][3] + 1e-8f;
        #pragma unroll
        for (int j = 0; j < NGATE; ++j) M[i][j] /= rs;
      }
    }
    #pragma unroll
    for (int i = 0; i < NGATE; ++i)
      s_res[t * NGATE + i] = M[i][0] + M[i][1] + M[i][2] + M[i][3];
  }
  __syncthreads();

  // ---- fused MLP via WMMA: each B fragment feeds BOTH M-tiles (register reuse) ----
  const int tbase = (lane & 16) >> 1;       // C/D fragment: row = vgpr_i + tbase
  const int cn    = lane & 15;              // C/D fragment: col = tile_base + cn

  v8f acc2[MT][8] = {};                     // GEMM2 accumulators (2 M-tiles x 128 cols)

  for (int kc = 0; kc < NCHUNK; ++kc) {
    // GEMM1: y[32 x 64] for this wave = xh(32x1024) @ W1^T columns [jb, jb+64)
    v8f acc1[MT][4] = {};
    const int jb = kc * KCH + wv * (KCH / 8);
    for (int kb = 0; kb < DIMC; kb += 32) {
      v16h a0 = load_a_lds(xh,              DIMC, lane, kb);
      v16h a1 = load_a_lds(xh + 16 * DIMC,  DIMC, lane, kb);
      #pragma unroll
      for (int nt = 0; nt < 4; ++nt) {
        v16h b = load_b_glb(W1h, DIMC, jb + nt * 16, kb, lane);
        acc1[0][nt] = __builtin_amdgcn_wmma_f32_16x16x32_f16(
            false, a0, false, b, (short)0, acc1[0][nt], false, false);
        acc1[1][nt] = __builtin_amdgcn_wmma_f32_16x16x32_f16(
            false, a1, false, b, (short)0, acc1[1][nt], false, false);
      }
    }
    // epilogue: h = gelu_exact(s_pre * y + b1) -> f16 chunk in LDS
    #pragma unroll
    for (int nt = 0; nt < 4; ++nt) {
      const int jl = wv * (KCH / 8) + nt * 16 + cn;    // column inside chunk
      const float bj = b1[kc * KCH + jl];
      #pragma unroll
      for (int m = 0; m < MT; ++m) {
        #pragma unroll
        for (int i = 0; i < 8; ++i) {
          const int t = m * 16 + tbase + i;
          const float y = fmaf(s_pre[t], acc1[m][nt][i], bj);
          const float g = 0.5f * y * (1.f + erff(y * 0.70710678118654752f));
          hch[t * KCH + jl] = (_Float16)g;
        }
      }
    }
    __syncthreads();

    // GEMM2 partial: acc2 += hch(32x512) @ W2^T[:, kc*512 .. ) for this wave's 128 cols
    const int nb0 = wv * (DIMC / 8);
    for (int kb = 0; kb < KCH; kb += 32) {
      v16h a0 = load_a_lds(hch,             KCH, lane, kb);
      v16h a1 = load_a_lds(hch + 16 * KCH,  KCH, lane, kb);
      #pragma unroll
      for (int nt = 0; nt < 8; ++nt) {
        v16h b = load_b_glb(W2h, HIDN, nb0 + nt * 16, kc * KCH + kb, lane);
        acc2[0][nt] = __builtin_amdgcn_wmma_f32_16x16x32_f16(
            false, a0, false, b, (short)0, acc2[0][nt], false, false);
        acc2[1][nt] = __builtin_amdgcn_wmma_f32_16x16x32_f16(
            false, a1, false, b, (short)0, acc2[1][nt], false, false);
      }
    }
    __syncthreads();                        // hch reused next chunk
  }

  // ---- output epilogue: out[t,n,c] = r[n]*x[c] + H_post[n]*(f_out[c]+b2[c]) ----
  #pragma unroll
  for (int nt = 0; nt < 8; ++nt) {
    const int c = wv * (DIMC / 8) + nt * 16 + cn;
    const float bc = b2[c];
    #pragma unroll
    for (int m = 0; m < MT; ++m) {
      #pragma unroll
      for (int i = 0; i < 8; ++i) {
        const int t = m * 16 + tbase + i;
        const size_t tok = tok0 + (size_t)t;
        const float fo = acc2[m][nt][i] + bc;
        const float xv = x[tok * DIMC + c];
        float* op = out + tok * (size_t)(NGATE * DIMC) + c;
        #pragma unroll
        for (int n = 0; n < NGATE; ++n)
          op[(size_t)n * DIMC] = fmaf(s_res[t * NGATE + n], xv, s_post[t * NGATE + n] * fo);
      }
    }
  }
}

// ====================================== launcher ======================================
extern "C" void kernel_launch(void* const* d_in, const int* in_sizes, int n_in,
                              void* d_out, int out_size, void* d_ws, size_t ws_size,
                              hipStream_t stream) {
  const float* x         = (const float*)d_in[0];
  const float* rms_w     = (const float*)d_in[1];
  const float* W_proj    = (const float*)d_in[2];
  const float* bias_pre  = (const float*)d_in[3];
  const float* bias_post = (const float*)d_in[4];
  const float* bias_res  = (const float*)d_in[5];
  const float* alpha_pre = (const float*)d_in[6];
  const float* alpha_post= (const float*)d_in[7];
  const float* alpha_res = (const float*)d_in[8];
  const float* W1        = (const float*)d_in[9];
  const float* b1        = (const float*)d_in[10];
  const float* W2        = (const float*)d_in[11];
  const float* b2        = (const float*)d_in[12];
  float* out = (float*)d_out;

  char* ws = (char*)d_ws;
  float*    W_eff = (float*)ws;                                       //  96 KB
  _Float16* W1h   = (_Float16*)(ws + 24 * DIMC * 4);                  //   8 MB
  _Float16* W2h   = (_Float16*)(ws + 24 * DIMC * 4 + (size_t)HIDN * DIMC * 2);

  mhc_prep<<<2048, 256, 0, stream>>>(W_proj, rms_w, W1, W2, W_eff, W1h, W2h);

  const int tokens = 4 * 2048;              // B * L
  mhc_main<<<tokens / NTOK, 256, 0, stream>>>(
      x, W_eff, bias_pre, bias_post, bias_res,
      alpha_pre, alpha_post, alpha_res,
      W1h, b1, W2h, b2, out);
}
